// GraphAttention_36584531427589
// MI455X (gfx1250) — matrix operand order repaired and checked by reading, not compiled, not run on hardware
//
#include <hip/hip_runtime.h>
#include <hip/hip_bf16.h>
#include <math.h>

#define HID   256
#define ATTN  192
#define NNODE 50000
#define KNBR  32

#define NODES_PER_WG 4
#define EROWS (NODES_PER_WG * KNBR)   // 128 edge rows per workgroup
#define HSTR  200                     // padded LDS row stride (bf16 elems) for h tile
#define ASTR  264                     // padded LDS row stride (bf16 elems) for emb tile
#define BSTR  200                     // padded LDS row stride (bf16) attn staged B
#define QBSTR 264                     // padded LDS row stride (bf16) qkv staged B
#define QROWS 128                     // node rows per qkv workgroup

#define USE_ASYNC_LDS 1               // gfx1250 async global->LDS path (inline asm)

typedef __attribute__((ext_vector_type(16))) __bf16 v16bf;
typedef __attribute__((ext_vector_type(8)))  float  v8f;

struct __align__(16) B128 { unsigned a, b, c, d; };
struct B128x2 { B128 lo, hi; };

static __device__ __forceinline__ v16bf make_frag(B128 lo, B128 hi) {
    B128x2 t{lo, hi};
    return __builtin_bit_cast(v16bf, t);
}

static __device__ __forceinline__ v16bf load_bfrag(const __bf16* p) {
    return make_frag(*(const B128*)p, *(const B128*)(p + 8));
}

// ---------------------------------------------------------------------------
// Phase 0: transpose + convert fp32 weight [K][C] -> bf16 WT [C][K]
// ---------------------------------------------------------------------------
__global__ void prep_wt_kernel(const float* __restrict__ W, __bf16* __restrict__ WT,
                               int K, int C) {
    int idx = blockIdx.x * blockDim.x + threadIdx.x;
    if (idx >= K * C) return;
    int c = idx / K, k = idx % K;
    WT[idx] = (__bf16)W[k * C + c];
}

// ---------------------------------------------------------------------------
// Phase 1: Q = emb@Wq, K = emb@Wk, V = emb@Wv  (WMMA bf16, f32 accumulate)
// One WG = 128 node rows (8 waves x one 16-row M-tile). All waves share each
// 16-col B tile, staged into double-buffered LDS via async global->LDS copies
// (ASYNCcnt) that overlap the previous tile's WMMAs.
// ---------------------------------------------------------------------------
__global__ __launch_bounds__(256)
void qkv_kernel(const float* __restrict__ emb,
                const __bf16* __restrict__ wqT,
                const __bf16* __restrict__ wkT,
                const __bf16* __restrict__ wvT,
                float* __restrict__ Qf,
                float* __restrict__ Kf,
                float* __restrict__ Vf) {
    __shared__ __align__(16) __bf16 aemb[QROWS * ASTR];      // ~67.6 KB
    __shared__ __align__(16) __bf16 qbstage[2 * 16 * QBSTR]; // ~16.9 KB

    const int t = threadIdx.x;
    const int w = t >> 5, lane = t & 31;
    const int nbase = blockIdx.x * QROWS;

    // stage 128 emb rows into LDS as bf16 (row-guarded for the tail block)
    for (int i = t; i < QROWS * HID; i += 256) {
        int r = i >> 8, c = i & 255;
        int grow = nbase + r;
        float x = (grow < NNODE) ? emb[(size_t)grow * HID + c] : 0.f;
        aemb[r * ASTR + c] = (__bf16)x;
    }

    auto wt_of = [&](int nt, const __bf16*& WT, int& cb) {
        if (nt < 12)      { WT = wqT; cb = nt * 16; }
        else if (nt < 24) { WT = wkT; cb = (nt - 12) * 16; }
        else              { WT = wvT; cb = (nt - 24) * 16; }
    };

    // stage one 16-row B tile (16 x 256 bf16 = 8 KB) into qbstage[buf]
    auto stage_tile = [&](int nt, int buf) {
#if USE_ASYNC_LDS
        if (w == 0) {
            const __bf16* WT; int cb;
            wt_of(nt, WT, cb);
            const __bf16* src = WT + (size_t)cb * HID;
            __builtin_prefetch(src + (size_t)32 * HID, 0, 3);   // ~two tiles ahead
#pragma unroll
            for (int i = 0; i < 16; ++i) {             // op i copies row i (32x16B)
                unsigned long long ga =
                    (unsigned long long)(const void*)(src + (size_t)i * HID + lane * 8);
                unsigned la = (unsigned)(unsigned long long)(const void*)
                    (qbstage + buf * (16 * QBSTR) + i * QBSTR + lane * 8);
                asm volatile("global_load_async_to_lds_b128 %0, %1, off"
                             :: "v"(la), "v"(ga) : "memory");
            }
        }
#else
        const __bf16* WT; int cb;
        wt_of(nt, WT, cb);
        const unsigned* src = (const unsigned*)(WT + (size_t)cb * HID);
        unsigned* dst = (unsigned*)(qbstage + buf * (16 * QBSTR));
        for (int idx = t; idx < 16 * 128; idx += 256) {  // 128 dwords per row
            int r = idx >> 7, c = idx & 127;
            dst[r * (QBSTR / 2) + c] = src[(size_t)r * (HID / 2) + c];
        }
#endif
    };
    auto wait_stage = [&]() {
#if USE_ASYNC_LDS
        if (w == 0)
            asm volatile("s_wait_asynccnt 0x0" ::: "memory");
#endif
    };

    stage_tile(0, 0);
    wait_stage();
    __syncthreads();            // A tile + staged B tile 0 visible

    const int m = lane & 15, kh = lane >> 4;
    const int mt = w;           // one 16-row M-tile per wave

    // hoist all 8 A fragments (invariant across the 40 col-tiles)
    v16bf afrag[8];
    {
        const __bf16* arow = aemb + (mt * 16 + m) * ASTR;
#pragma unroll
        for (int kk = 0; kk < 8; ++kk) {
            B128 alo = *(const B128*)(arow + kk * 32 + kh * 8);
            B128 ahi = *(const B128*)(arow + kk * 32 + 16 + kh * 8);
            afrag[kk] = make_frag(alo, ahi);
        }
    }

    for (int nt = 0; nt < 40; ++nt) {
        int buf = nt & 1;
        if (nt + 1 < 40)
            stage_tile(nt + 1, buf ^ 1);   // async copy overlaps this tile's math

        const __bf16* brow = qbstage + buf * (16 * QBSTR) + m * QBSTR + kh * 8;

        v8f c = {};
#pragma unroll
        for (int kk = 0; kk < 8; ++kk) {
            v16bf b = make_frag(*(const B128*)(brow + kk * 32), *(const B128*)(brow + kk * 32 + 16));      // ds_load_b128 x2
            c = __builtin_amdgcn_wmma_f32_16x16x32_bf16(false, afrag[kk], false, b,
                                                        (short)0, c, false, false);
        }

        const __bf16* WT; int cb;
        wt_of(nt, WT, cb);
        float* Out = (nt < 12) ? Qf : (nt < 24) ? Kf : Vf;
        int    OC  = (nt < 24) ? ATTN : HID;

        int row0 = nbase + mt * 16 + 8 * kh;
        float* p = Out + (size_t)row0 * OC + cb + m;
#pragma unroll
        for (int v = 0; v < 8; ++v)
            if (row0 + v < NNODE)
                p[(size_t)v * OC] = c[v];

        wait_stage();           // drain ASYNCcnt (wave 0), then block-wide hand-off
        __syncthreads();
    }
}

// ---------------------------------------------------------------------------
// Phase 2: fused edge-MLP (WMMA) + scores + softmax + V aggregation + residual
// One WG = 4 nodes = 128 edge rows. W2 N-tiles staged via async global->LDS
// double buffering (unchanged from the round that compiled).
// ---------------------------------------------------------------------------
__global__ __launch_bounds__(256)
void attn_kernel(const float* __restrict__ emb,
                 const int*   __restrict__ nbr_idx,
                 const float* __restrict__ xy,
                 const float* __restrict__ W1,
                 const float* __restrict__ b1,
                 const float* __restrict__ b2,
                 const __bf16* __restrict__ w2T,
                 const float* __restrict__ Qf,
                 const float* __restrict__ Kf,
                 const float* __restrict__ Vf,
                 float* __restrict__ out) {
    __shared__ __align__(16) __bf16 h_lds[EROWS * HSTR];       // ~50 KB
    __shared__ __align__(16) __bf16 bstage[2 * 16 * BSTR];     // ~12.5 KB
    __shared__ float sc[EROWS];
    __shared__ float wgt[EROWS];
    __shared__ int   nbr_s[EROWS];
    __shared__ float qs[NODES_PER_WG * ATTN];

    const int t = threadIdx.x;
    const int w = t >> 5, lane = t & 31;
    const int nodeBase = blockIdx.x * NODES_PER_WG;

    for (int i = t; i < NODES_PER_WG * ATTN; i += 256)
        qs[i] = Qf[(size_t)(nodeBase + i / ATTN) * ATTN + (i % ATTN)];

    // ---- Step A: edge features + layer-1 relu -> LDS bf16 (2 threads/row) ----
    {
        int row  = t >> 1;
        int half = t & 1;
        int node = nodeBase + (row >> 5);
        int kk   = row & 31;
        int nb   = nbr_idx[(size_t)node * KNBR + kk];
        if (half == 0) nbr_s[row] = nb;
        float dx = xy[2 * (size_t)nb + 0] - xy[2 * (size_t)node + 0];
        float dy = xy[2 * (size_t)nb + 1] - xy[2 * (size_t)node + 1];
        float r  = sqrtf(dx * dx + dy * dy);
        float sn, cs;
        if (r > 0.f) { float inv = 1.f / r; sn = dy * inv; cs = dx * inv; }
        else         { sn = 0.f; cs = 1.f; }            // atan2(0,0)=0 -> sin 0, cos 1
        int a0 = half * (ATTN / 2);
        for (int a = a0; a < a0 + ATTN / 2; ++a) {
            float h = fmaf(r, W1[a],
                      fmaf(sn, W1[ATTN + a],
                      fmaf(cs, W1[2 * ATTN + a], b1[a])));
            h_lds[row * HSTR + a] = (__bf16)fmaxf(h, 0.f);
        }
    }

    auto stage_tile = [&](int nt, int buf) {
#if USE_ASYNC_LDS
        if (w == 0) {
            const __bf16* src = w2T + (size_t)(nt * 16) * ATTN;
#pragma unroll
            for (int i = 0; i < 12; ++i) {               // 12 x 32 lanes x 16B = 6 KB
                int q = i * 32 + lane;
                int r = q / 24, c16 = q % 24;            // 24 16B-chunks per 384B row
                unsigned long long ga =
                    (unsigned long long)(const void*)(src + r * ATTN + c16 * 8);
                unsigned la = (unsigned)(unsigned long long)(const void*)
                    (bstage + buf * (16 * BSTR) + r * BSTR + c16 * 8);
                asm volatile("global_load_async_to_lds_b128 %0, %1, off"
                             :: "v"(la), "v"(ga) : "memory");
            }
        }
#else
        const unsigned* src = (const unsigned*)(w2T + (size_t)(nt * 16) * ATTN);
        unsigned* dst = (unsigned*)(bstage + buf * (16 * BSTR));
        for (int idx = t; idx < 16 * 96; idx += 256) {   // 96 dwords per row
            int r = idx / 96, c = idx % 96;
            dst[r * (BSTR / 2) + c] = src[(size_t)r * (ATTN / 2) + c];
        }
#endif
    };
    auto wait_stage = [&]() {
#if USE_ASYNC_LDS
        if (w == 0)
            asm volatile("s_wait_asynccnt 0x0" ::: "memory");
#endif
    };

    stage_tile(0, 0);
    wait_stage();
    __syncthreads();

    // ---- Step B: E = h @ W2 (+b2), fused with score = Q . (K[nbr] + E) ----
    const int n = lane & 15, kh = lane >> 4;
    const int mt = w;

    int nbv[8];
    const float* qrow[8];
#pragma unroll
    for (int v = 0; v < 8; ++v) {
        int row = mt * 16 + 8 * kh + v;
        nbv[v]  = nbr_s[row];
        qrow[v] = qs + (row >> 5) * ATTN;
    }

    v16bf afrag[6];
    {
        const __bf16* arow = h_lds + (mt * 16 + n) * HSTR;
#pragma unroll
        for (int kk = 0; kk < 6; ++kk) {
            B128 alo = *(const B128*)(arow + kk * 32 + kh * 8);
            B128 ahi = *(const B128*)(arow + kk * 32 + 16 + kh * 8);
            afrag[kk] = make_frag(alo, ahi);
        }
    }

    float rs[8];
#pragma unroll
    for (int v = 0; v < 8; ++v) rs[v] = 0.f;

    for (int nt = 0; nt < 12; ++nt) {
        int buf = nt & 1;
        if (nt + 1 < 12)
            stage_tile(nt + 1, buf ^ 1);

        int cb = nt * 16;
        const __bf16* brow = bstage + buf * (16 * BSTR) + n * BSTR + kh * 8;

        v8f c = {};
#pragma unroll
        for (int kk = 0; kk < 6; ++kk) {
            v16bf b = make_frag(*(const B128*)(brow + kk * 32), *(const B128*)(brow + kk * 32 + 16));
            c = __builtin_amdgcn_wmma_f32_16x16x32_bf16(false, afrag[kk], false, b,
                                                        (short)0, c, false, false);
        }

        float b2v = b2[cb + n];
#pragma unroll
        for (int v = 0; v < 8; ++v) {
            float e = c[v] + b2v + Kf[(size_t)nbv[v] * ATTN + cb + n];
            float contrib = qrow[v][cb + n] * e;
            contrib += __shfl_xor(contrib, 1, 32);
            contrib += __shfl_xor(contrib, 2, 32);
            contrib += __shfl_xor(contrib, 4, 32);
            contrib += __shfl_xor(contrib, 8, 32);
            rs[v] += contrib;
        }

        wait_stage();
        __syncthreads();
    }
    if (n == 0) {
#pragma unroll
        for (int v = 0; v < 8; ++v)
            sc[mt * 16 + 8 * kh + v] = rs[v];
    }
    __syncthreads();

    // ---- Step C: softmax over 32 neighbors (one wave per node) ----
    const float scale = 0.0721687836f;                  // 1/sqrt(192)
    if (t < EROWS) {
        float s  = sc[t] * scale;
        float mx = s;
        for (int d = 1; d < 32; d <<= 1) mx = fmaxf(mx, __shfl_xor(mx, d, 32));
        float e = __expf(s - mx);
        float sum = e;
        for (int d = 1; d < 32; d <<= 1) sum += __shfl_xor(sum, d, 32);
        wgt[t] = e / sum;
    }
    __syncthreads();

    // ---- Step D: agg = sum_k w * V[nbr], residual add (coalesced over HID) ----
#pragma unroll
    for (int j = 0; j < NODES_PER_WG; ++j) {
        float acc = 0.f;
        const float* wj  = wgt + j * KNBR;
        const int*   nbj = nbr_s + j * KNBR;
        for (int k = 0; k < KNBR; ++k)
            acc = fmaf(wj[k], Vf[(size_t)nbj[k] * HID + t], acc);
        size_t o = (size_t)(nodeBase + j) * HID + t;
        out[o] = emb[o] + acc;
    }
}

// ---------------------------------------------------------------------------
extern "C" void kernel_launch(void* const* d_in, const int* in_sizes, int n_in,
                              void* d_out, int out_size, void* d_ws, size_t ws_size,
                              hipStream_t stream) {
    const float* emb = (const float*)d_in[0];
    const int*   nbr = (const int*)  d_in[1];
    const float* xy  = (const float*)d_in[2];
    const float* Wq  = (const float*)d_in[3];
    const float* Wk  = (const float*)d_in[4];
    const float* Wv  = (const float*)d_in[5];
    const float* W1  = (const float*)d_in[6];
    const float* b1  = (const float*)d_in[7];
    const float* W2  = (const float*)d_in[8];
    const float* b2  = (const float*)d_in[9];
    float* out = (float*)d_out;

    char* ws = (char*)d_ws;
    size_t off = 0;
    auto alloc = [&](size_t bytes) -> void* {
        void* p = ws + off;
        off = (off + bytes + 255) & ~(size_t)255;
        return p;
    };
    __bf16* wqT = (__bf16*)alloc((size_t)ATTN * HID * sizeof(__bf16));
    __bf16* wkT = (__bf16*)alloc((size_t)ATTN * HID * sizeof(__bf16));
    __bf16* wvT = (__bf16*)alloc((size_t)HID  * HID * sizeof(__bf16));
    __bf16* w2T = (__bf16*)alloc((size_t)ATTN * ATTN * sizeof(__bf16));
    float*  Qf  = (float*) alloc((size_t)NNODE * ATTN * sizeof(float));
    float*  Kf  = (float*) alloc((size_t)NNODE * ATTN * sizeof(float));
    float*  Vf  = (float*) alloc((size_t)NNODE * HID  * sizeof(float));

    prep_wt_kernel<<<(HID * ATTN + 255) / 256, 256, 0, stream>>>(Wq, wqT, HID, ATTN);
    prep_wt_kernel<<<(HID * ATTN + 255) / 256, 256, 0, stream>>>(Wk, wkT, HID, ATTN);
    prep_wt_kernel<<<(HID * HID  + 255) / 256, 256, 0, stream>>>(Wv, wvT, HID, HID);
    prep_wt_kernel<<<(ATTN * ATTN + 255) / 256, 256, 0, stream>>>(W2, w2T, ATTN, ATTN);

    qkv_kernel<<<(NNODE + QROWS - 1) / QROWS, 256, 0, stream>>>(
        emb, wqT, wkT, wvT, Qf, Kf, Vf);

    attn_kernel<<<NNODE / NODES_PER_WG, 256, 0, stream>>>(
        emb, nbr, xy, W1, b1, b2, w2T, Qf, Kf, Vf, out);
}
